// GraphSAGE_90117003805342
// MI455X (gfx1250) — matrix operand order repaired and checked by reading, not compiled, not run on hardware
//
#include <hip/hip_runtime.h>
#include <hip/hip_bf16.h>

// ---------- types ----------
typedef __bf16 bf16_t;
typedef __attribute__((ext_vector_type(16))) __bf16       v16bf;
typedef __attribute__((ext_vector_type(8)))  float        v8f;
typedef __attribute__((ext_vector_type(4)))  unsigned int u32x4;
typedef __attribute__((ext_vector_type(8)))  unsigned int u32x8;
typedef __attribute__((ext_vector_type(4)))  float        f32x4;

__device__ __forceinline__ bf16_t f2bf(float f) {
  unsigned u = __builtin_bit_cast(unsigned, f);
  u += 0x7FFFu + ((u >> 16) & 1u);                 // round-to-nearest-even
  unsigned short h = (unsigned short)(u >> 16);
  return __builtin_bit_cast(bf16_t, h);
}
__device__ __forceinline__ unsigned short bfbits(float f) {
  return __builtin_bit_cast(unsigned short, f2bf(f));
}
__device__ __forceinline__ float bfraw2f(unsigned short h) {
  unsigned u = ((unsigned)h) << 16;
  return __builtin_bit_cast(float, u);
}

// ---------- prep kernels ----------
// Transpose + convert weight f32 [K][N] -> bf16 Bt [N][Ktot] at row offset koff.
__global__ void k_wtrans(const float* __restrict__ W, bf16_t* __restrict__ Bt,
                         int K, int N, int Ktot, int koff) {
  int i = blockIdx.x * blockDim.x + threadIdx.x;
  if (i >= K * N) return;
  int k = i / N, n = i % N;
  Bt[(long)n * Ktot + koff + k] = f2bf(W[i]);
}

// Out[r][c] = bf16(X[r][c] * nm[r]) with output row stride ldo (cols 0..D-1).
__global__ void k_maskcvt(const float* __restrict__ X, const int* __restrict__ nm,
                          bf16_t* __restrict__ Out, int D, int ldo, long total) {
  long i = (long)blockIdx.x * blockDim.x + threadIdx.x;
  if (i >= total) return;
  long r = i / D; int c = (int)(i % D);
  Out[r * (long)ldo + c] = f2bf(X[i] * (float)nm[r]);
}

// ---------- aggregation (f32 children, D=128), one wave per node ----------
// Out[node][coloff + d] = sum_f nbm*nmC*X[node*16+f][d] / max(sum_f nbm, 1)
__global__ void k_agg_f32_d128(const float* __restrict__ X, const int* __restrict__ nbm,
                               const int* __restrict__ nmC, bf16_t* __restrict__ Out,
                               int ldo, int coloff) {
  const int wid = threadIdx.x >> 5, lane = threadIdx.x & 31;
  const long node = (long)blockIdx.x * 8 + wid;
  f32x4 s = (f32x4)0.f;
  float cnt = 0.f;
  const float* base = X + node * 16 * 128 + lane * 4;
  #pragma unroll
  for (int f = 0; f < 16; ++f) {
    int nb = nbm[node * 16 + f];           // wave-uniform
    cnt += (float)nb;
    float w = (float)(nb * nmC[node * 16 + f]);
    if (w != 0.f) {                        // uniform branch: skips dead-neighbor reads
      f32x4 v = *(const f32x4*)(base + f * 128);
      s += v * w;
    }
  }
  float inv = 1.f / fmaxf(cnt, 1.f);
  s *= inv;
  bf16_t* o = Out + node * ldo + coloff + lane * 4;
  o[0] = f2bf(s[0]); o[1] = f2bf(s[1]); o[2] = f2bf(s[2]); o[3] = f2bf(s[3]);
}

// ---------- aggregation (bf16 children, D=256), one wave per node ----------
__global__ void k_agg_bf16_d256(const bf16_t* __restrict__ X, const int* __restrict__ nbm,
                                bf16_t* __restrict__ Out, int ldo, int coloff) {
  const int wid = threadIdx.x >> 5, lane = threadIdx.x & 31;
  const long node = (long)blockIdx.x * 8 + wid;
  float s[8];
  #pragma unroll
  for (int d = 0; d < 8; ++d) s[d] = 0.f;
  float cnt = 0.f;
  const bf16_t* base = X + node * 16 * 256 + lane * 8;
  #pragma unroll
  for (int f = 0; f < 16; ++f) {
    int nb = nbm[node * 16 + f];
    cnt += (float)nb;
    if (nb) {
      u32x4 q = *(const u32x4*)(base + f * 256);
      #pragma unroll
      for (int d = 0; d < 4; ++d) {
        s[2 * d]     += bfraw2f((unsigned short)(q[d] & 0xFFFFu));
        s[2 * d + 1] += bfraw2f((unsigned short)(q[d] >> 16));
      }
    }
  }
  float inv = 1.f / fmaxf(cnt, 1.f);
  u32x4 o;
  #pragma unroll
  for (int d = 0; d < 4; ++d) {
    unsigned lo = bfbits(s[2 * d] * inv);
    unsigned hi = bfbits(s[2 * d + 1] * inv);
    o[d] = lo | (hi << 16);
  }
  *(u32x4*)(Out + node * ldo + coloff + lane * 8) = o;
}

// ---------- WMMA GEMM ----------
// C[M x BN] = A[M x K] (bf16, row-major) * Bt[BN x K] (bf16, K contiguous per row)
// epilogue: + bias0[n] (+ bias1[n]) ; optional ReLU ; * rowmask[row] ; store bf16/f32.
// Block: 256 threads = 8 waves; wave w owns rows [blk*128 + w*16, +16), all BN cols.
//
// Pipelining:
//  - Slab level: next K-slab of Bt and next A fragment are fetched from global
//    into registers while the current slab is consumed from LDS (hides the
//    global fetch behind 16 WMMAs).
//  - Tile level: B fragments are 2-deep double buffered so ds_load_b128 pairs
//    issue one WMMA ahead (partial s_wait_dscnt instead of wait-to-zero).
template<int BN, bool RELU, bool OUT_F32>
__launch_bounds__(256)
__global__ void k_gemm(const bf16_t* __restrict__ A, int K,
                       const bf16_t* __restrict__ Bt,
                       const float* __restrict__ bias0, const float* __restrict__ bias1,
                       const int* __restrict__ rowmask,
                       void* __restrict__ Cout, int ldc) {
  constexpr int NT = BN / 16;
  constexpr int LDS_ROW = 80;                 // 64B of K-slab + 16B pad (bank spread)
  constexpr int CH = (BN * 4) / 256;          // 16B chunks per thread per slab
  __shared__ __align__(16) unsigned char ldsB[BN * LDS_ROW];

  const int tid   = threadIdx.x;
  const int wid   = tid >> 5;
  const int lane  = tid & 31;
  const int lhalf = lane >> 4;                // 0: K-low half, 1: K-high half
  const int l16   = lane & 15;
  const long m0   = (long)blockIdx.x * 128 + wid * 16;

  v8f acc[NT];
  #pragma unroll
  for (int t = 0; t < NT; ++t) acc[t] = (v8f)0.f;

  const bf16_t* arow = A + (m0 + l16) * K + lhalf * 16;

  // ---- prologue: stage slab 0 (Bt) and A fragment 0 into registers ----
  u32x4 stage[CH];
  #pragma unroll
  for (int c = 0; c < CH; ++c) {
    int idx = tid + c * 256;
    int n = idx >> 2, part = idx & 3;
    stage[c] = *(const u32x4*)(Bt + (long)n * K + part * 8);
  }
  u32x4 alo = *(const u32x4*)(arow);
  u32x4 ahi = *(const u32x4*)(arow + 8);

  for (int kb = 0; kb < K; kb += 32) {
    const bool more = (kb + 32) < K;

    __syncthreads();                          // prior compute done reading LDS
    // commit staged slab to LDS
    #pragma unroll
    for (int c = 0; c < CH; ++c) {
      int idx = tid + c * 256;
      int n = idx >> 2, part = idx & 3;
      *(u32x4*)(&ldsB[n * LDS_ROW + part * 16]) = stage[c];
    }

    // current A fragment registers
    u32x4 a0 = alo, a1 = ahi;

    // issue global fetches for the NEXT slab + next A fragment now; they
    // complete while this slab's 16 WMMAs run.
    if (more) {
      #pragma unroll
      for (int c = 0; c < CH; ++c) {
        int idx = tid + c * 256;
        int n = idx >> 2, part = idx & 3;
        stage[c] = *(const u32x4*)(Bt + (long)n * K + (kb + 32) + part * 8);
      }
      alo = *(const u32x4*)(arow + kb + 32);
      ahi = *(const u32x4*)(arow + kb + 40);
      if (kb + 64 < K) __builtin_prefetch(arow + kb + 64, 0, 0);  // global_prefetch_b8
    }

    __syncthreads();                          // slab visible to all waves

    v16bf afrag = __builtin_bit_cast(
        v16bf, __builtin_shufflevector(a0, a1, 0, 1, 2, 3, 4, 5, 6, 7));

    // 2-deep pipelined B fragments from LDS
    u32x4 blo[2], bhi[2];
    {
      const unsigned char* bp = &ldsB[l16 * LDS_ROW + lhalf * 32];
      blo[0] = *(const u32x4*)bp;
      bhi[0] = *(const u32x4*)(bp + 16);
    }
    #pragma unroll
    for (int t = 0; t < NT; ++t) {
      if (t + 1 < NT) {
        const unsigned char* bp = &ldsB[((t + 1) * 16 + l16) * LDS_ROW + lhalf * 32];
        blo[(t + 1) & 1] = *(const u32x4*)bp;
        bhi[(t + 1) & 1] = *(const u32x4*)(bp + 16);
      }
      v16bf bfrag = __builtin_bit_cast(
          v16bf, __builtin_shufflevector(blo[t & 1], bhi[t & 1],
                                         0, 1, 2, 3, 4, 5, 6, 7));
      acc[t] = __builtin_amdgcn_wmma_f32_16x16x32_bf16(
          false, afrag, false, bfrag, (short)0, acc[t], false, false);
    }
  }

  // epilogue: C element (t, j) of this lane is C[m0 + 8*lhalf + j][t*16 + l16]
  float rm[8];
  #pragma unroll
  for (int j = 0; j < 8; ++j) {
    long row = m0 + lhalf * 8 + j;
    rm[j] = rowmask ? (float)rowmask[row] : 1.f;
  }
  #pragma unroll
  for (int t = 0; t < NT; ++t) {
    int n = t * 16 + l16;
    float bv = (bias0 ? bias0[n] : 0.f) + (bias1 ? bias1[n] : 0.f);
    #pragma unroll
    for (int j = 0; j < 8; ++j) {
      long row = m0 + lhalf * 8 + j;
      float v = acc[t][j] + bv;
      if (RELU) v = fmaxf(v, 0.f);
      v *= rm[j];
      if constexpr (OUT_F32) ((float*)Cout)[row * (long)ldc + n] = v;
      else                   ((bf16_t*)Cout)[row * (long)ldc + n] = f2bf(v);
    }
  }
}

// ---------- launch ----------
extern "C" void kernel_launch(void* const* d_in, const int* in_sizes, int n_in,
                              void* d_out, int out_size, void* d_ws, size_t ws_size,
                              hipStream_t stream) {
  const float* x0      = (const float*)d_in[0];
  const float* x1      = (const float*)d_in[1];
  const float* x2      = (const float*)d_in[2];
  const float* W_self0 = (const float*)d_in[3];
  const float* b_self0 = (const float*)d_in[4];
  const float* W_nei0  = (const float*)d_in[5];
  const float* b_nei0  = (const float*)d_in[6];
  const float* W_self1 = (const float*)d_in[7];
  const float* b_self1 = (const float*)d_in[8];
  const float* W_nei1  = (const float*)d_in[9];
  const float* b_nei1  = (const float*)d_in[10];
  const float* W_cls   = (const float*)d_in[11];
  const float* b_cls   = (const float*)d_in[12];
  const int*   nm0     = (const int*)d_in[13];
  const int*   nm1     = (const int*)d_in[14];
  const int*   nm2     = (const int*)d_in[15];
  const int*   nbm0    = (const int*)d_in[16];
  const int*   nbm1    = (const int*)d_in[17];

  constexpr long N0 = 4096, N1 = 65536;

  unsigned char* ws = (unsigned char*)d_ws;
  size_t off = 0;
  auto alloc = [&](size_t bytes) -> void* {
    void* p = ws + off;
    off = (off + bytes + 255) & ~(size_t)255;
    return p;
  };
  bf16_t* A1  = (bf16_t*)alloc(N1 * 256 * 2);   // [s1 | agg1], K=256
  bf16_t* H1  = (bf16_t*)alloc(N1 * 256 * 2);   // layer0 hidden of level-1 nodes
  bf16_t* A0  = (bf16_t*)alloc(N0 * 256 * 2);   // [s0 | agg0], K=256
  bf16_t* A2  = (bf16_t*)alloc(N0 * 512 * 2);   // [h0 | agg(h1)], K=512
  bf16_t* S3  = (bf16_t*)alloc(N0 * 256 * 2);   // layer1 output state
  bf16_t* Bt0 = (bf16_t*)alloc(256 * 256 * 2);  // [Wself0;Wnei0]^T
  bf16_t* Bt1 = (bf16_t*)alloc(256 * 512 * 2);  // [Wself1;Wnei1]^T
  bf16_t* Btc = (bf16_t*)alloc(64  * 256 * 2);  // Wcls^T

  // --- weight prep (transpose to [N][K] bf16, concat along K) ---
  k_wtrans<<<(128 * 256 + 255) / 256, 256, 0, stream>>>(W_self0, Bt0, 128, 256, 256, 0);
  k_wtrans<<<(128 * 256 + 255) / 256, 256, 0, stream>>>(W_nei0,  Bt0, 128, 256, 256, 128);
  k_wtrans<<<(256 * 256 + 255) / 256, 256, 0, stream>>>(W_self1, Bt1, 256, 256, 512, 0);
  k_wtrans<<<(256 * 256 + 255) / 256, 256, 0, stream>>>(W_nei1,  Bt1, 256, 256, 512, 256);
  k_wtrans<<<(256 * 64  + 255) / 256, 256, 0, stream>>>(W_cls,   Btc, 256, 64,  256, 0);

  // --- masked+bf16 node features into concat buffers (cols 0..127) ---
  k_maskcvt<<<(int)((N1 * 128 + 255) / 256), 256, 0, stream>>>(x1, nm1, A1, 128, 256, N1 * 128);
  k_maskcvt<<<(int)((N0 * 128 + 255) / 256), 256, 0, stream>>>(x0, nm0, A0, 128, 256, N0 * 128);

  // --- aggregations over f32 children (cols 128..255) ---
  k_agg_f32_d128<<<(int)(N1 / 8), 256, 0, stream>>>(x2, nbm1, nm2, A1, 256, 128);
  k_agg_f32_d128<<<(int)(N0 / 8), 256, 0, stream>>>(x1, nbm0, nm1, A0, 256, 128);

  // --- layer 0: h = relu([s|agg] @ [Ws;Wn] + bs + bn) * nm ---
  k_gemm<256, true, false><<<(int)(N1 / 128), 256, 0, stream>>>(
      A1, 256, Bt0, b_self0, b_nei0, nm1, H1, 256);
  k_gemm<256, true, false><<<(int)(N0 / 128), 256, 0, stream>>>(
      A0, 256, Bt0, b_self0, b_nei0, nm0, A2, 512);   // h0 -> A2 cols 0..255

  // --- layer 1 aggregation of h1 (bf16) -> A2 cols 256..511 ---
  k_agg_bf16_d256<<<(int)(N0 / 8), 256, 0, stream>>>(H1, nbm0, A2, 512, 256);

  // --- layer 1: s = ([h0|agg] @ [Ws1;Wn1] + bs1 + bn1) * nm0 (no relu) ---
  k_gemm<256, false, false><<<(int)(N0 / 128), 256, 0, stream>>>(
      A2, 512, Bt1, b_self1, b_nei1, nm0, S3, 256);

  // --- classifier: logits = s @ Wcls + bcls (f32 out) ---
  k_gemm<64, false, true><<<(int)(N0 / 128), 256, 0, stream>>>(
      S3, 256, Btc, b_cls, nullptr, nullptr, d_out, 64);
}
